// SonicMoEAdapter_49185965473857
// MI455X (gfx1250) — matrix operand order repaired and checked by reading, not compiled, not run on hardware
//
#include <hip/hip_runtime.h>
#include <hip/hip_bf16.h>
#include <cstdint>

#define TOK  4096
#define DM   1024
#define DFF  2048
#define NEXP 8

typedef __attribute__((ext_vector_type(16))) __bf16 v16bf;
typedef __attribute__((ext_vector_type(8)))  __bf16 v8bf;
typedef __attribute__((ext_vector_type(4)))  __bf16 v4bf;
typedef __attribute__((ext_vector_type(8)))  float  v8f;

// round-to-nearest-even f32 -> bf16 (manual, no bf16 arithmetic required)
__device__ __forceinline__ __bf16 f2bf(float f) {
  union { float f; uint32_t u; } v; v.f = f;
  uint32_t r = v.u + 0x7FFFu + ((v.u >> 16) & 1u);
  uint16_t s = (uint16_t)(r >> 16);
  return __builtin_bit_cast(__bf16, s);
}

// A fragment (16x32 bf16): lane = 16*half + m ; contiguous K in [16h, 16h+16)
__device__ __forceinline__ v16bf loadA(const __bf16* __restrict__ row, int k, int half) {
  return *(const v16bf*)(row + k + half * 16);
}

// B fragment (32x16 bf16): lane = 16*half + n ; contiguous K in [16h, 16h+16)
// reading row n of the row-major [N,K] weight (computes X * W^T)
__device__ __forceinline__ v16bf loadB(const __bf16* __restrict__ row, int k, int half) {
  return *(const v16bf*)(row + k + half * 16);
}

#define WMMA_BF16(A, B, C) \
  __builtin_amdgcn_wmma_f32_16x16x32_bf16(false, (A), false, (B), (short)0, (C), false, false)

// ---------------------------------------------------------------- init
__global__ __launch_bounds__(256) void k_init(float* __restrict__ y,
                                              int* __restrict__ counts) {
  size_t i = (size_t)blockIdx.x * 256 + threadIdx.x;   // 4096 blocks * 256 * float4 = T*D
  float4 z = make_float4(0.f, 0.f, 0.f, 0.f);
  ((float4*)y)[i] = z;
  if (blockIdx.x == 0 && threadIdx.x < NEXP) counts[threadIdx.x] = 0;
}

// ---------------------------------------------------------------- f32 -> bf16
__global__ __launch_bounds__(256) void k_cvt(const float* __restrict__ s,
                                             __bf16* __restrict__ d, int n4) {
  int stride = gridDim.x * blockDim.x;
  for (int i = blockIdx.x * blockDim.x + threadIdx.x; i < n4; i += stride) {
    float4 v = ((const float4*)s)[i];
    v4bf o;
    o[0] = f2bf(v.x); o[1] = f2bf(v.y); o[2] = f2bf(v.z); o[3] = f2bf(v.w);
    ((v4bf*)d)[i] = o;
  }
}

// ---------------------------------------------------------------- router (wave32 per token)
__global__ __launch_bounds__(256) void k_router(
    const float* __restrict__ x, const float* __restrict__ wr,
    float* __restrict__ probs, int* __restrict__ idx_list,
    float* __restrict__ wgt_list, int* __restrict__ counts)
{
  const int lane = threadIdx.x & 31;
  const int wave = threadIdx.x >> 5;
  const int t = blockIdx.x * 8 + wave;

  float acc[NEXP];
#pragma unroll
  for (int e = 0; e < NEXP; ++e) acc[e] = 0.f;
  const float* xr = x + (size_t)t * DM;
  for (int i = lane; i < DM; i += 32) {
    float xv = xr[i];
#pragma unroll
    for (int e = 0; e < NEXP; ++e) acc[e] += xv * wr[e * DM + i];
  }
#pragma unroll
  for (int off = 16; off > 0; off >>= 1) {
#pragma unroll
    for (int e = 0; e < NEXP; ++e) acc[e] += __shfl_xor(acc[e], off, 32);
  }
  if (lane == 0) {
    float mx = acc[0];
#pragma unroll
    for (int e = 1; e < NEXP; ++e) mx = fmaxf(mx, acc[e]);
    float p[NEXP]; float s = 0.f;
#pragma unroll
    for (int e = 0; e < NEXP; ++e) { p[e] = __expf(acc[e] - mx); s += p[e]; }
    float inv = 1.f / s;
#pragma unroll
    for (int e = 0; e < NEXP; ++e) { p[e] *= inv; probs[t * NEXP + e] = p[e]; }
    // top-2 (first-occurrence tie break like jax top_k)
    int i0 = 0;
#pragma unroll
    for (int e = 1; e < NEXP; ++e) if (p[e] > p[i0]) i0 = e;
    int i1 = (i0 == 0) ? 1 : 0;
#pragma unroll
    for (int e = 0; e < NEXP; ++e) if (e != i0 && p[e] > p[i1]) i1 = e;
    float w0 = p[i0], w1 = p[i1];
    float rs = 1.f / (w0 + w1);
    w0 *= rs; w1 *= rs;
    int s0 = atomicAdd(&counts[i0], 1);
    idx_list[i0 * TOK + s0] = t * 2;       // packed = token*2 + kslot
    wgt_list[i0 * TOK + s0] = w0;
    int s1 = atomicAdd(&counts[i1], 1);
    idx_list[i1 * TOK + s1] = t * 2 + 1;
    wgt_list[i1 * TOK + s1] = w1;
  }
}

// ---------------------------------------------------------------- aux loss (deterministic)
__global__ __launch_bounds__(256) void k_aux(const float* __restrict__ probs,
                                             const int* __restrict__ counts,
                                             float* __restrict__ out_aux) {
  __shared__ float red[256];
  const int tid = threadIdx.x;
  float acc[NEXP];
#pragma unroll
  for (int e = 0; e < NEXP; ++e) acc[e] = 0.f;
  for (int t = tid; t < TOK; t += 256) {
#pragma unroll
    for (int e = 0; e < NEXP; ++e) acc[e] += probs[t * NEXP + e];
  }
  float aux = 0.f;
  for (int e = 0; e < NEXP; ++e) {
    red[tid] = acc[e];
    __syncthreads();
    for (int s = 128; s > 0; s >>= 1) {
      if (tid < s) red[tid] += red[tid + s];
      __syncthreads();
    }
    if (tid == 0) {
      float P = red[0] * (1.f / (float)TOK);
      float f = (float)counts[e] * (1.f / (float)TOK);
      aux += f * P;
    }
    __syncthreads();
  }
  if (tid == 0) *out_aux = (float)NEXP * aux * 0.01f;
}

// ---------------------------------------------------------------- gate/up GEMM -> h
// block tile 128(M) x 64(FF-cols); 8 waves: M-pair p = w&3 (rows p*32..p*32+31),
// col half = w>>2 (2 tiles). Per wave/K-step: 2 A + 4 B frags -> 8 WMMAs.
__global__ __launch_bounds__(256) void k_gateup(
    const __bf16* __restrict__ xb, const __bf16* __restrict__ wg,
    const __bf16* __restrict__ wu, const int* __restrict__ idx_list,
    const int* __restrict__ counts, __bf16* __restrict__ h)
{
  const int wave = threadIdx.x >> 5, lane = threadIdx.x & 31;
  const int e = blockIdx.x >> 5, rowTile = blockIdx.x & 31;
  const int cnt = counts[e];
  const int mBase = rowTile * 128;
  if (mBase >= cnt) return;

  const int p = wave & 3;
  const int nBase = blockIdx.y * 64 + (wave >> 2) * 32;
  const int r = lane & 15, half = lane >> 4;

  const int m0 = mBase + p * 32 + r;
  const int m1 = m0 + 16;
  const int pk0 = idx_list[e * TOK + (m0 < cnt ? m0 : cnt - 1)];
  const int pk1 = idx_list[e * TOK + (m1 < cnt ? m1 : cnt - 1)];
  const __bf16* arow0 = xb + (size_t)(pk0 >> 1) * DM;
  const __bf16* arow1 = xb + (size_t)(pk1 >> 1) * DM;
  const __bf16* bg0 = wg + (size_t)e * DFF * DM + (size_t)(nBase + r) * DM;
  const __bf16* bg1 = bg0 + 16 * DM;
  const __bf16* bu0 = wu + (size_t)e * DFF * DM + (size_t)(nBase + r) * DM;
  const __bf16* bu1 = bu0 + 16 * DM;

  v8f z = {0.f,0.f,0.f,0.f,0.f,0.f,0.f,0.f};
  v8f cg00 = z, cg01 = z, cu00 = z, cu01 = z;   // A0 x {g0,g1,u0,u1}
  v8f cg10 = z, cg11 = z, cu10 = z, cu11 = z;   // A1 x {g0,g1,u0,u1}

  // prologue fragments (K=0)
  v16bf a0 = loadA(arow0, 0, half);
  v16bf a1 = loadA(arow1, 0, half);
  v16bf g0 = loadB(bg0, 0, half);
  v16bf g1 = loadB(bg1, 0, half);
  v16bf u0 = loadB(bu0, 0, half);
  v16bf u1 = loadB(bu1, 0, half);

#pragma unroll 2
  for (int k = 32; k < DM; k += 32) {
    v16bf a0n = loadA(arow0, k, half);
    v16bf a1n = loadA(arow1, k, half);
    v16bf g0n = loadB(bg0, k, half);
    v16bf g1n = loadB(bg1, k, half);
    v16bf u0n = loadB(bu0, k, half);
    v16bf u1n = loadB(bu1, k, half);
    cg00 = WMMA_BF16(a0, g0, cg00);
    cg01 = WMMA_BF16(a0, g1, cg01);
    cu00 = WMMA_BF16(a0, u0, cu00);
    cu01 = WMMA_BF16(a0, u1, cu01);
    cg10 = WMMA_BF16(a1, g0, cg10);
    cg11 = WMMA_BF16(a1, g1, cg11);
    cu10 = WMMA_BF16(a1, u0, cu10);
    cu11 = WMMA_BF16(a1, u1, cu11);
    a0 = a0n; a1 = a1n; g0 = g0n; g1 = g1n; u0 = u0n; u1 = u1n;
  }
  cg00 = WMMA_BF16(a0, g0, cg00);
  cg01 = WMMA_BF16(a0, g1, cg01);
  cu00 = WMMA_BF16(a0, u0, cu00);
  cu01 = WMMA_BF16(a0, u1, cu01);
  cg10 = WMMA_BF16(a1, g0, cg10);
  cg11 = WMMA_BF16(a1, g1, cg11);
  cu10 = WMMA_BF16(a1, u0, cu10);
  cu11 = WMMA_BF16(a1, u1, cu11);

  // epilogue: silu(gate)*up -> bf16 h. D layout: m = i + 8*half, n = r
#pragma unroll
  for (int i = 0; i < 8; ++i) {
    int mA = mBase + p * 32 + half * 8 + i;        // A0 block rows
    if (mA < cnt) {
      int pk = idx_list[e * TOK + mA];
      float g = cg00[i], u = cu00[i];
      h[(size_t)pk * DFF + nBase + r] = f2bf((g / (1.f + __expf(-g))) * u);
      g = cg01[i]; u = cu01[i];
      h[(size_t)pk * DFF + nBase + 16 + r] = f2bf((g / (1.f + __expf(-g))) * u);
    }
    int mB = mA + 16;                              // A1 block rows
    if (mB < cnt) {
      int pk = idx_list[e * TOK + mB];
      float g = cg10[i], u = cu10[i];
      h[(size_t)pk * DFF + nBase + r] = f2bf((g / (1.f + __expf(-g))) * u);
      g = cg11[i]; u = cu11[i];
      h[(size_t)pk * DFF + nBase + 16 + r] = f2bf((g / (1.f + __expf(-g))) * u);
    }
  }
}

// ---------------------------------------------------------------- down GEMM -> scatter y
// block tile 128(M) x 128(D-cols); 8 waves: M-pair p = w&3, col half = w>>2 (4 tiles)
// Per wave/K-step: 2 A + 4 B frags -> 8 WMMAs.
__global__ __launch_bounds__(256) void k_down(
    const __bf16* __restrict__ h, const __bf16* __restrict__ wd,
    const int* __restrict__ idx_list, const float* __restrict__ wgt_list,
    const int* __restrict__ counts, float* __restrict__ y)
{
  const int wave = threadIdx.x >> 5, lane = threadIdx.x & 31;
  const int e = blockIdx.x >> 5, rowTile = blockIdx.x & 31;
  const int cnt = counts[e];
  const int mBase = rowTile * 128;
  if (mBase >= cnt) return;

  const int p = wave & 3;
  const int nBase = blockIdx.y * 128 + (wave >> 2) * 64;
  const int r = lane & 15, half = lane >> 4;

  const int m0 = mBase + p * 32 + r;
  const int m1 = m0 + 16;
  const int pk0 = idx_list[e * TOK + (m0 < cnt ? m0 : cnt - 1)];
  const int pk1 = idx_list[e * TOK + (m1 < cnt ? m1 : cnt - 1)];
  const __bf16* arow0 = h + (size_t)pk0 * DFF;
  const __bf16* arow1 = h + (size_t)pk1 * DFF;
  const __bf16* b0 = wd + (size_t)e * DM * DFF + (size_t)(nBase + r) * DFF;
  const __bf16* b1 = b0 + 16 * DFF;
  const __bf16* b2 = b0 + 32 * DFF;
  const __bf16* b3 = b0 + 48 * DFF;

  v8f z = {0.f,0.f,0.f,0.f,0.f,0.f,0.f,0.f};
  v8f c00 = z, c01 = z, c02 = z, c03 = z;   // A0 x b0..b3
  v8f c10 = z, c11 = z, c12 = z, c13 = z;   // A1 x b0..b3

  v16bf a0  = loadA(arow0, 0, half);
  v16bf a1  = loadA(arow1, 0, half);
  v16bf bb0 = loadB(b0, 0, half);
  v16bf bb1 = loadB(b1, 0, half);
  v16bf bb2 = loadB(b2, 0, half);
  v16bf bb3 = loadB(b3, 0, half);

#pragma unroll 2
  for (int k = 32; k < DFF; k += 32) {
    v16bf a0n  = loadA(arow0, k, half);
    v16bf a1n  = loadA(arow1, k, half);
    v16bf bb0n = loadB(b0, k, half);
    v16bf bb1n = loadB(b1, k, half);
    v16bf bb2n = loadB(b2, k, half);
    v16bf bb3n = loadB(b3, k, half);
    c00 = WMMA_BF16(a0, bb0, c00);
    c01 = WMMA_BF16(a0, bb1, c01);
    c02 = WMMA_BF16(a0, bb2, c02);
    c03 = WMMA_BF16(a0, bb3, c03);
    c10 = WMMA_BF16(a1, bb0, c10);
    c11 = WMMA_BF16(a1, bb1, c11);
    c12 = WMMA_BF16(a1, bb2, c12);
    c13 = WMMA_BF16(a1, bb3, c13);
    a0 = a0n; a1 = a1n; bb0 = bb0n; bb1 = bb1n; bb2 = bb2n; bb3 = bb3n;
  }
  c00 = WMMA_BF16(a0, bb0, c00);
  c01 = WMMA_BF16(a0, bb1, c01);
  c02 = WMMA_BF16(a0, bb2, c02);
  c03 = WMMA_BF16(a0, bb3, c03);
  c10 = WMMA_BF16(a1, bb0, c10);
  c11 = WMMA_BF16(a1, bb1, c11);
  c12 = WMMA_BF16(a1, bb2, c12);
  c13 = WMMA_BF16(a1, bb3, c13);

#pragma unroll
  for (int i = 0; i < 8; ++i) {
    int mA = mBase + p * 32 + half * 8 + i;
    if (mA < cnt) {
      int pk = idx_list[e * TOK + mA];
      float w = wgt_list[e * TOK + mA];
      float* yr = y + (size_t)(pk >> 1) * DM + nBase + r;
      atomicAdd(yr,      c00[i] * w);
      atomicAdd(yr + 16, c01[i] * w);
      atomicAdd(yr + 32, c02[i] * w);
      atomicAdd(yr + 48, c03[i] * w);
    }
    int mB = mA + 16;
    if (mB < cnt) {
      int pk = idx_list[e * TOK + mB];
      float w = wgt_list[e * TOK + mB];
      float* yr = y + (size_t)(pk >> 1) * DM + nBase + r;
      atomicAdd(yr,      c10[i] * w);
      atomicAdd(yr + 16, c11[i] * w);
      atomicAdd(yr + 32, c12[i] * w);
      atomicAdd(yr + 48, c13[i] * w);
    }
  }
}

// ---------------------------------------------------------------- launch
extern "C" void kernel_launch(void* const* d_in, const int* in_sizes, int n_in,
                              void* d_out, int out_size, void* d_ws, size_t ws_size,
                              hipStream_t stream) {
  (void)in_sizes; (void)n_in; (void)out_size; (void)ws_size;
  const float* x  = (const float*)d_in[0];
  const float* wr = (const float*)d_in[1];
  const float* wg = (const float*)d_in[2];
  const float* wu = (const float*)d_in[3];
  const float* wd = (const float*)d_in[4];
  float* y   = (float*)d_out;
  float* aux = y + (size_t)TOK * DM;

  char* ws = (char*)d_ws;
  size_t off = 0;
  auto alloc = [&](size_t bytes) -> char* {
    char* p = ws + off;
    off += (bytes + 255) & ~(size_t)255;
    return p;
  };
  __bf16* xb   = (__bf16*)alloc((size_t)TOK * DM * 2);
  __bf16* wgb  = (__bf16*)alloc((size_t)NEXP * DFF * DM * 2);
  __bf16* wub  = (__bf16*)alloc((size_t)NEXP * DFF * DM * 2);
  __bf16* wdb  = (__bf16*)alloc((size_t)NEXP * DM * DFF * 2);
  __bf16* h    = (__bf16*)alloc((size_t)2 * TOK * DFF * 2);
  float* probs    = (float*)alloc((size_t)TOK * NEXP * 4);
  int*   idx_list = (int*)alloc((size_t)NEXP * TOK * 4);
  float* wgt_list = (float*)alloc((size_t)NEXP * TOK * 4);
  int*   counts   = (int*)alloc(NEXP * 4);

  k_init<<<4096, 256, 0, stream>>>(y, counts);
  k_cvt<<<2048, 256, 0, stream>>>(x,  xb,  TOK * DM / 4);
  k_cvt<<<2048, 256, 0, stream>>>(wg, wgb, NEXP * DFF * DM / 4);
  k_cvt<<<2048, 256, 0, stream>>>(wu, wub, NEXP * DFF * DM / 4);
  k_cvt<<<2048, 256, 0, stream>>>(wd, wdb, NEXP * DM * DFF / 4);
  k_router<<<TOK / 8, 256, 0, stream>>>(x, wr, probs, idx_list, wgt_list, counts);
  k_aux<<<1, 256, 0, stream>>>(probs, counts, aux);
  dim3 gA(NEXP * 32, DFF / 64);
  k_gateup<<<gA, 256, 0, stream>>>(xb, wgb, wub, idx_list, counts, h);
  dim3 gB(NEXP * 32, DM / 128);
  k_down<<<gB, 256, 0, stream>>>(h, wdb, idx_list, wgt_list, counts, y);
}
